// NOTEARS_21492016349912
// MI455X (gfx1250) — matrix operand order repaired and checked by reading, not compiled, not run on hardware
//
#include <hip/hip_runtime.h>
#include <hip/hip_bf16.h>

// ---------------------------------------------------------------------------
// Types for CDNA5 WMMA intrinsics (wave32)
// ---------------------------------------------------------------------------
typedef __bf16 bf16_t;
typedef bf16_t v16bf __attribute__((ext_vector_type(16)));
typedef bf16_t v2bf  __attribute__((ext_vector_type(2)));
typedef float  v8f   __attribute__((ext_vector_type(8)));
typedef float  v4f   __attribute__((ext_vector_type(4)));
typedef float  v2f   __attribute__((ext_vector_type(2)));

#define D        128
#define CHUNK    64          // rows of X per LDS tile (2 x K=32 WMMA steps)
#define USTR     72          // ushort stride of transposed tile row: 144 B, 16B-aligned
#define USTR32   36          // same stride in uint (bf16-pair) units
#define GRAM_BLOCKS 512
#define STR      129         // f32 LDS stride for d x d matrices (odd dwords)

#if __has_builtin(__builtin_amdgcn_wmma_f32_16x16x4_f32)
#define HAVE_WMMA_F32 1
#else
#define HAVE_WMMA_F32 0
#endif

// round-to-nearest-even fp32 -> bf16 (low 16 bits)
__device__ __forceinline__ unsigned int f2bf(float f) {
  unsigned u = __float_as_uint(f);
  u += 0x7FFFu + ((u >> 16) & 1u);
  return u >> 16;
}

// pack two fp32 -> {lo,hi} bf16 pair; prefer hw v_cvt_pk_bf16_f32
__device__ __forceinline__ unsigned int pack2bf(float fa, float fb) {
#if __has_builtin(__builtin_amdgcn_cvt_pk_bf16_f32)
  v2bf p = __builtin_amdgcn_cvt_pk_bf16_f32(fa, fb);
  return __builtin_bit_cast(unsigned int, p);
#else
  return f2bf(fa) | (f2bf(fb) << 16);
#endif
}

union ABPack { v16bf v; uint4 q[2]; };

// ---------------------------------------------------------------------------
// Kernel 0: zero the Gram accumulator in workspace (ws is poisoned by harness)
// ---------------------------------------------------------------------------
__global__ void zero_kernel(float* __restrict__ p, int nElem) {
  int i = blockIdx.x * blockDim.x + threadIdx.x;
  if (i < nElem) p[i] = 0.0f;
}

// ---------------------------------------------------------------------------
// Gram compute step over one 64-row chunk: 2 K-halves x 8 column tiles
// = 16 v_wmma_f32_16x16x32_bf16 per wave.
// ---------------------------------------------------------------------------
__device__ __forceinline__ void gram_compute(const unsigned short* buf,
                                             v8f acc[8], int w, int n, int hi) {
#pragma unroll
  for (int kh = 0; kh < 2; ++kh) {
    const int kb = kh * 32;
    // A (16x32 bf16): half0 VGPR0-3=K0..7, VGPR4-7=K16..23; half1: K8..15,K24..31
    ABPack A;
    const int arow = (w * 16 + n) * USTR + kb;
    A.q[0] = *reinterpret_cast<const uint4*>(&buf[arow + (hi ? 8 : 0)]);
    A.q[1] = *reinterpret_cast<const uint4*>(&buf[arow + (hi ? 24 : 16)]);
    // B (32x16 bf16): half0 K0..7,K16..23; half1 K8..15,K24..31
#pragma unroll
    for (int jt = 0; jt < 8; ++jt) {
      ABPack B;
      const int brow = (jt * 16 + n) * USTR + kb + (hi ? 8 : 0);
      B.q[0] = *reinterpret_cast<const uint4*>(&buf[brow]);
      B.q[1] = *reinterpret_cast<const uint4*>(&buf[brow + 16]);
      acc[jt] = __builtin_amdgcn_wmma_f32_16x16x32_bf16(
          false, A.v, false, B.v, (short)0, acc[jt], false, false);
    }
  }
}

// ---------------------------------------------------------------------------
// Kernel 1: G = X^T X via bf16 WMMA (fp32 accumulate). HBM-bound: reads X once
// (non-temporal), double-buffered LDS with one barrier/iteration, next chunk's
// global loads issued before the barrier so they overlap this chunk's WMMAs.
// ---------------------------------------------------------------------------
__global__ __launch_bounds__(256) void gram_bf16_kernel(
    const float* __restrict__ X, float* __restrict__ G, int T) {
  // transposed chunk, bf16 pairs: uint at [c*USTR32 + tpair] = {t=2tp lo, 2tp+1 hi}
  __shared__ unsigned int sXT[2][D * USTR32];

  const int tid  = threadIdx.x;
  const int w    = tid >> 5;        // wave id 0..7 (wave32)
  const int lane = tid & 31;
  const int n    = lane & 15;
  const int hi   = lane >> 4;
  // load mapping: thread owns rows {2rp, 2rp+1, 2rp+32, 2rp+33}, 8-col block cb
  const int rp   = tid >> 4;        // 0..15
  const int cb   = tid & 15;        // 0..15

  const long long stride_rows = (long long)gridDim.x * CHUNK;
  v8f acc[8] = {};                  // full 16x128 slab of G in registers
  v4f a0, a1, b0, b1, c0, c1, d0, d1;

  long long t0 = (long long)blockIdx.x * CHUNK;
  bool have = (t0 + CHUNK <= T);
  if (have) {
    const v4f* pa = reinterpret_cast<const v4f*>(X + (size_t)(t0 + 2 * rp) * D + cb * 8);
    const v4f* pb = reinterpret_cast<const v4f*>(X + (size_t)(t0 + 2 * rp + 1) * D + cb * 8);
    const v4f* pc = reinterpret_cast<const v4f*>(X + (size_t)(t0 + 2 * rp + 32) * D + cb * 8);
    const v4f* pd = reinterpret_cast<const v4f*>(X + (size_t)(t0 + 2 * rp + 33) * D + cb * 8);
    a0 = __builtin_nontemporal_load(pa); a1 = __builtin_nontemporal_load(pa + 1);
    b0 = __builtin_nontemporal_load(pb); b1 = __builtin_nontemporal_load(pb + 1);
    c0 = __builtin_nontemporal_load(pc); c1 = __builtin_nontemporal_load(pc + 1);
    d0 = __builtin_nontemporal_load(pd); d1 = __builtin_nontemporal_load(pd + 1);
  }

  int p = 0;
  while (have) {
    // pack fp32->bf16 pairs (v_cvt_pk_bf16_f32) and store transposed
    // (16 x ds_store_b32 per thread)
#pragma unroll
    for (int i = 0; i < 8; ++i) {
      float fa = (i < 4) ? a0[i] : a1[i - 4];
      float fb = (i < 4) ? b0[i] : b1[i - 4];
      float fc = (i < 4) ? c0[i] : c1[i - 4];
      float fd = (i < 4) ? d0[i] : d1[i - 4];
      const int c = cb * 8 + i;
      sXT[p][c * USTR32 + rp]      = pack2bf(fa, fb);  // rows 2rp, 2rp+1
      sXT[p][c * USTR32 + 16 + rp] = pack2bf(fc, fd);  // rows 2rp+32, 2rp+33
    }
    // issue next chunk's global loads now; they complete under this chunk's WMMAs
    const long long t1 = t0 + stride_rows;
    const bool have_next = (t1 + CHUNK <= T);
    if (have_next) {
      const v4f* pa = reinterpret_cast<const v4f*>(X + (size_t)(t1 + 2 * rp) * D + cb * 8);
      const v4f* pb = reinterpret_cast<const v4f*>(X + (size_t)(t1 + 2 * rp + 1) * D + cb * 8);
      const v4f* pc = reinterpret_cast<const v4f*>(X + (size_t)(t1 + 2 * rp + 32) * D + cb * 8);
      const v4f* pd = reinterpret_cast<const v4f*>(X + (size_t)(t1 + 2 * rp + 33) * D + cb * 8);
      a0 = __builtin_nontemporal_load(pa); a1 = __builtin_nontemporal_load(pa + 1);
      b0 = __builtin_nontemporal_load(pb); b1 = __builtin_nontemporal_load(pb + 1);
      c0 = __builtin_nontemporal_load(pc); c1 = __builtin_nontemporal_load(pc + 1);
      d0 = __builtin_nontemporal_load(pd); d1 = __builtin_nontemporal_load(pd + 1);
    }
    __syncthreads();   // single barrier: double buffering makes this sufficient
    gram_compute(reinterpret_cast<const unsigned short*>(&sXT[p][0]), acc, w, n, hi);
    p ^= 1;
    t0 = t1;
    have = have_next;
  }

  // tail chunk (block-uniform condition; only if T % CHUNK spills for this block)
  if (t0 < T) {
#pragma unroll
    for (int i = 0; i < 8; ++i) {
      const int c = cb * 8 + i;
      unsigned u0 = 0, u1 = 0, u2 = 0, u3 = 0;
      if (t0 + 2 * rp < T)      u0 = f2bf(X[(size_t)(t0 + 2 * rp) * D + c]);
      if (t0 + 2 * rp + 1 < T)  u1 = f2bf(X[(size_t)(t0 + 2 * rp + 1) * D + c]);
      if (t0 + 2 * rp + 32 < T) u2 = f2bf(X[(size_t)(t0 + 2 * rp + 32) * D + c]);
      if (t0 + 2 * rp + 33 < T) u3 = f2bf(X[(size_t)(t0 + 2 * rp + 33) * D + c]);
      sXT[p][c * USTR32 + rp]      = u0 | (u1 << 16);
      sXT[p][c * USTR32 + 16 + rp] = u2 | (u3 << 16);
    }
    __syncthreads();
    gram_compute(reinterpret_cast<const unsigned short*>(&sXT[p][0]), acc, w, n, hi);
  }

  // merge partial G into global accumulator (hw fp32 atomics)
  // D layout: lane n(+16*hi), VGPR v <-> (M = v + 8*hi, N = n) within tile
#pragma unroll
  for (int jt = 0; jt < 8; ++jt) {
#pragma unroll
    for (int v = 0; v < 8; ++v) {
      int row = w * 16 + v + 8 * hi;
      int col = jt * 16 + n;
      unsafeAtomicAdd(&G[row * D + col], acc[jt][v]);
    }
  }
}

// ---------------------------------------------------------------------------
// Kernel 2 helpers: 128x128 fp32 matmul slab per wave via v_wmma_f32_16x16x4_f32
// ---------------------------------------------------------------------------
__device__ __forceinline__ float block_reduce(float v, float* red, int tid) {
  red[tid] = v;
  __syncthreads();
#pragma unroll
  for (int s = 128; s > 0; s >>= 1) {
    if (tid < s) red[tid] += red[tid + s];
    __syncthreads();
  }
  float r = red[0];
  __syncthreads();
  return r;
}

// slab = A(rows 16w..16w+15) @ B into acc[8] (D-matrix register layout)
__device__ __forceinline__ void matmul_slab(const float* sA, const float* sB,
                                            int w, int n, int hi, v8f acc[8]) {
  v8f z = {};
#pragma unroll
  for (int jt = 0; jt < 8; ++jt) acc[jt] = z;
#if HAVE_WMMA_F32
  const int arow = (16 * w + n) * STR;
  for (int k4 = 0; k4 < 32; ++k4) {
    int ka = 4 * k4 + 2 * hi;           // f32 A: half0 K{0,1}, half1 K{2,3}
    v2f a;
    a.x = sA[arow + ka];
    a.y = sA[arow + ka + 1];
#pragma unroll
    for (int jt = 0; jt < 8; ++jt) {
      v2f b;                            // f32 B: half0 rows{0,1}, half1 rows{2,3}
      b.x = sB[ka * STR + 16 * jt + n];
      b.y = sB[(ka + 1) * STR + 16 * jt + n];
      acc[jt] = __builtin_amdgcn_wmma_f32_16x16x4_f32(
          false, a, false, b, (short)0, acc[jt], false, false);
    }
  }
#else
#pragma unroll
  for (int jt = 0; jt < 8; ++jt) {
#pragma unroll
    for (int v = 0; v < 8; ++v) {
      int M = 16 * w + v + 8 * hi, N = 16 * jt + n;
      float s = 0.f;
      for (int k = 0; k < D; ++k) s += sA[M * STR + k] * sB[k * STR + N];
      acc[jt][v] = s;
    }
  }
#endif
}

__device__ __forceinline__ void store_slab(float* sDst, int w, int n, int hi,
                                           const v8f acc[8]) {
#pragma unroll
  for (int jt = 0; jt < 8; ++jt)
#pragma unroll
    for (int v = 0; v < 8; ++v)
      sDst[(16 * w + v + 8 * hi) * STR + 16 * jt + n] = acc[jt][v];
}

// ---------------------------------------------------------------------------
// Kernel 2: assemble full NOTEARS objective from G and W (single workgroup).
// loss = 0.5/n*(trG - 2*sum(G o Wm) + sum((Wm G) o Wm)) + 0.5h + 0.5h^2 + 0.01*L1
// h    = sum_{k=1..9} tr((Wm o Wm)^k) / k!
// Uses 2 x 66 KB fp32 LDS matrices (CDNA5 WGP has 320 KB LDS).
// ---------------------------------------------------------------------------
__global__ __launch_bounds__(256) void notears_finalize(
    const float* __restrict__ W, const float* __restrict__ G,
    float* __restrict__ out, float invn) {
  __shared__ float sWW[D * STR];   // holds Wm, then WW = Wm o Wm
  __shared__ float sP[D * STR];    // holds G, then power chain
  __shared__ float sRed[256];

  const int tid  = threadIdx.x;
  const int w    = tid >> 5;
  const int lane = tid & 31;
  const int n    = lane & 15;
  const int hi   = lane >> 4;

  // Phase 0: stage Wm and G to LDS; accumulate trG, s1 = sum(G o Wm), L1
  float trG = 0.f, s1 = 0.f, l1 = 0.f;
  for (int it = 0; it < 64; ++it) {
    int idx = tid + it * 256;
    int i = idx >> 7, j = idx & 127;
    float g  = G[idx];
    float wv = (i == j) ? 0.f : W[idx];
    sWW[i * STR + j] = wv;
    sP[i * STR + j]  = g;
    s1 += g * wv;
    l1 += fabsf(wv);
    if (i == j) trG += g;
  }
  __syncthreads();
  trG = block_reduce(trG, sRed, tid);
  s1  = block_reduce(s1,  sRed, tid);
  l1  = block_reduce(l1,  sRed, tid);

  // Phase 1: s2 = sum over (Wm @ G) o Wm   (one 128^3 fp32 WMMA matmul)
  v8f acc[8];
  matmul_slab(sWW, sP, w, n, hi, acc);
  float s2 = 0.f;
#pragma unroll
  for (int jt = 0; jt < 8; ++jt)
#pragma unroll
    for (int v = 0; v < 8; ++v)
      s2 += acc[jt][v] * sWW[(16 * w + v + 8 * hi) * STR + 16 * jt + n];
  s2 = block_reduce(s2, sRed, tid);

  // Phase 2: WW = Wm o Wm into both buffers (each element owned by one thread)
  for (int it = 0; it < 64; ++it) {
    int idx = tid + it * 256;
    int i = idx >> 7, j = idx & 127;
    float wv = sWW[i * STR + j];
    float ww = wv * wv;
    sWW[i * STR + j] = ww;
    sP[i * STR + j]  = ww;
  }
  __syncthreads();

  // Phase 3: h = sum_{k=1..9} tr(WW^k)/k!   (power chain, 8 matmuls)
  float tloc = (tid < D) ? sP[tid * STR + tid] : 0.f;
  float htot = block_reduce(tloc, sRed, tid);   // k = 1, /1!
  float factorial = 1.f;
  for (int k = 2; k <= 9; ++k) {
    matmul_slab(sP, sWW, w, n, hi, acc);        // power = power @ WW
    __syncthreads();
    store_slab(sP, w, n, hi, acc);
    __syncthreads();
    factorial *= (float)k;
    tloc = (tid < D) ? sP[tid * STR + tid] : 0.f;
    htot += block_reduce(tloc, sRed, tid) / factorial;
  }

  if (tid == 0) {
    float loss = 0.5f * invn * (trG - 2.f * s1 + s2);
    out[0] = loss + 0.5f * htot + 0.5f * htot * htot + 0.01f * l1;
  }
}

// ---------------------------------------------------------------------------
// Host launcher
// ---------------------------------------------------------------------------
extern "C" void kernel_launch(void* const* d_in, const int* in_sizes, int n_in,
                              void* d_out, int out_size, void* d_ws, size_t ws_size,
                              hipStream_t stream) {
  (void)n_in; (void)out_size; (void)ws_size;
  const float* X = (const float*)d_in[0];
  const float* W = (const float*)d_in[1];
  const int T = in_sizes[0] / D;       // 1,000,000
  float* G = (float*)d_ws;             // 128*128 fp32 accumulator

  zero_kernel<<<(D * D + 255) / 256, 256, 0, stream>>>(G, D * D);
  gram_bf16_kernel<<<GRAM_BLOCKS, 256, 0, stream>>>(X, G, T);
  notears_finalize<<<1, 256, 0, stream>>>(W, G, (float*)d_out, 1.0f / (float)T);
}
